// GraphTransformerLayer_52707838656823
// MI455X (gfx1250) — compile-verified, matrix operand order checked by reading, not run on hardware
//
#include <hip/hip_runtime.h>
#include <hip/hip_bf16.h>
#include <math.h>

// ---- problem constants (from reference) ----
constexpr int N_NODES = 20000;
constexpr int E_EDGES = 320000;
constexpr int IN_CH   = 128;
constexpr int DH      = 64;     // per-head dim / out dim
constexpr int HEADS   = 4;
constexpr int HD      = HEADS * DH;      // 256
constexpr int COLS    = 3 * HD + DH;     // 832 fused projection columns: [Q|K|V|skip]
constexpr int CTILES  = COLS / 16;       // 52 column tiles of 16
constexpr int OFF_Q   = 0;
constexpr int OFF_K   = HD;              // 256
constexpr int OFF_V   = 2 * HD;          // 512
constexpr int OFF_S   = 3 * HD;          // 768

typedef __bf16 v16bf __attribute__((ext_vector_type(16)));
typedef __bf16 v8bf  __attribute__((ext_vector_type(8)));
typedef float  v8f   __attribute__((ext_vector_type(8)));

union ABFrag { v16bf v; v8bf h[2]; };

// order-preserving encode of float into uint for atomicMax (0 == -inf identity)
__device__ __forceinline__ unsigned encf(float f) {
    unsigned u = __float_as_uint(f);
    return (u & 0x80000000u) ? ~u : (u | 0x80000000u);
}
__device__ __forceinline__ float decf(unsigned u) {
    return __uint_as_float((u & 0x80000000u) ? (u ^ 0x80000000u) : ~u);
}

// ---- k0: zero-init accumulators (msg, denom, smax) ----
__global__ void k_init(float* __restrict__ msg, float* __restrict__ denom,
                       unsigned* __restrict__ smax) {
    int i = blockIdx.x * blockDim.x + threadIdx.x;
    if (i < N_NODES * HD) msg[i] = 0.f;
    if (i < N_NODES * HEADS) { denom[i] = 0.f; smax[i] = 0u; }
}

// ---- k1: convert x (f32 [N,128]) -> bf16 ----
__global__ void k_conv_x(const float* __restrict__ x, __bf16* __restrict__ xb) {
    int i = blockIdx.x * blockDim.x + threadIdx.x;
    if (i < N_NODES * IN_CH) xb[i] = (__bf16)x[i];
}

// ---- k2: build fused weight (bf16, col-major [COLS][128]) + fused bias ----
__global__ void k_build_w(const float* __restrict__ Wq, const float* __restrict__ Wk,
                          const float* __restrict__ Wv, const float* __restrict__ Ws,
                          const float* __restrict__ bq, const float* __restrict__ bk,
                          const float* __restrict__ bv, const float* __restrict__ bs,
                          __bf16* __restrict__ Wall, float* __restrict__ biasAll) {
    int i = blockIdx.x * blockDim.x + threadIdx.x;
    if (i >= COLS * IN_CH) return;
    int c = i / IN_CH, k = i % IN_CH;
    float w;
    if      (c < OFF_K) w = Wq[(size_t)k * HD + c];
    else if (c < OFF_V) w = Wk[(size_t)k * HD + (c - OFF_K)];
    else if (c < OFF_S) w = Wv[(size_t)k * HD + (c - OFF_V)];
    else                w = Ws[(size_t)k * DH + (c - OFF_S)];
    Wall[(size_t)c * IN_CH + k] = (__bf16)w;
    if (k == 0) {
        float b = (c < OFF_K) ? bq[c] : (c < OFF_V) ? bk[c - OFF_K]
                : (c < OFF_S) ? bv[c - OFF_V] : bs[c - OFF_S];
        biasAll[c] = b;
    }
}

// ---- k3: fused Q/K/V/skip projection GEMM via bf16 WMMA ----
// grid: N/16 blocks, 128 threads (4 waves). Each wave: 16x16 tile, loops 13 col tiles.
// Output goes to one fused buffer P[N][832] -> branchless 8x global_store_b32 epilogue.
__global__ void __launch_bounds__(128)
k_gemm(const __bf16* __restrict__ xb, const __bf16* __restrict__ Wall,
       const float* __restrict__ biasAll, float* __restrict__ P) {
    const int row0 = blockIdx.x * 16;
    const int lane = threadIdx.x & 31;
    const int wave = threadIdx.x >> 5;
    const int m    = lane & 15;
    const int kb   = (lane >> 4) << 3;   // 0 for lanes 0-15, 8 for lanes 16-31

    // A fragments (x rows, bf16, 16x32 per k-step), loaded once, reused for all tiles
    const v8bf* xr = (const v8bf*)(xb + (size_t)(row0 + m) * IN_CH);
    ABFrag afr[4];
#pragma unroll
    for (int kk = 0; kk < 4; ++kk) {
        const int ks = kk * 32;
        afr[kk].h[0] = xr[(ks + kb) >> 3];        // K = ks+kb .. +7
        afr[kk].h[1] = xr[(ks + 16 + kb) >> 3];   // K = ks+16+kb .. +7
    }

    const int n     = lane & 15;
    const int rbase = row0 + ((lane >> 4) << 3); // M = vgpr + (hi-half ? 8 : 0)

    for (int t = wave; t < CTILES; t += 4) {
        const int c = t * 16 + n;                // this lane's output column (B: N=lane&15)
        const v8bf* wc = (const v8bf*)(Wall + (size_t)c * IN_CH);

        // prefetch next tile's weight column into cache while WMMAs run
        if (t + 4 < CTILES)
            __builtin_prefetch(Wall + (size_t)(c + 64) * IN_CH, 0, 1);

        v8f acc = {};
#pragma unroll
        for (int kk = 0; kk < 4; ++kk) {
            const int ks = kk * 32;
            ABFrag bfr;
            bfr.h[0] = wc[(ks + kb) >> 3];
            bfr.h[1] = wc[(ks + 16 + kb) >> 3];
            acc = __builtin_amdgcn_wmma_f32_16x16x32_bf16(
                false, afr[kk].v, false, bfr.v, (short)0, acc, false, false);
        }
        const float bias = biasAll[c];
        float* db = P + (size_t)rbase * COLS + c;
#pragma unroll
        for (int i = 0; i < 8; ++i)
            db[(size_t)i * COLS] = acc[i] + bias;   // row stride 3328B -> imm offsets
    }
}

// ---- k4: per-(edge,head) attention logits + segment max ----
__global__ void k_scores(const int* __restrict__ ei, const float* __restrict__ P,
                         float* __restrict__ scores, unsigned* __restrict__ smax) {
    int idx = blockIdx.x * blockDim.x + threadIdx.x;
    if (idx >= E_EDGES * HEADS) return;
    const int e = idx >> 2, h = idx & 3;
    const int s = ei[e], d = ei[E_EDGES + e];
    const float4* qr = (const float4*)(P + (size_t)d * COLS + OFF_Q + h * DH);
    const float4* kr = (const float4*)(P + (size_t)s * COLS + OFF_K + h * DH);
    float acc = 0.f;
#pragma unroll
    for (int i = 0; i < 16; ++i) {
        float4 a = qr[i], b = kr[i];
        acc = fmaf(a.x, b.x, acc); acc = fmaf(a.y, b.y, acc);
        acc = fmaf(a.z, b.z, acc); acc = fmaf(a.w, b.w, acc);
    }
    const float sc = acc * 0.125f;  // 1/sqrt(64)
    scores[idx] = sc;
    atomicMax(&smax[d * HEADS + h], encf(sc));
}

// ---- k5: exp + unnormalized aggregation (msg += p*v, denom += p) ----
__global__ void k_agg(const int* __restrict__ ei, const float* __restrict__ scores,
                      const unsigned* __restrict__ smax, const float* __restrict__ P,
                      float* __restrict__ denom, float* __restrict__ msg) {
    int idx = blockIdx.x * blockDim.x + threadIdx.x;
    if (idx >= E_EDGES * HEADS) return;
    const int e = idx >> 2, h = idx & 3;
    const int s = ei[e], d = ei[E_EDGES + e];
    const float p = __expf(scores[idx] - decf(smax[d * HEADS + h]));
    atomicAdd(&denom[d * HEADS + h], p);
    const float4* vr = (const float4*)(P + (size_t)s * COLS + OFF_V + h * DH);
    float* mr = msg + (size_t)d * HD + h * DH;
#pragma unroll
    for (int i = 0; i < 16; ++i) {
        float4 v = vr[i];
        atomicAdd(mr + i * 4 + 0, p * v.x);
        atomicAdd(mr + i * 4 + 1, p * v.y);
        atomicAdd(mr + i * 4 + 2, p * v.z);
        atomicAdd(mr + i * 4 + 3, p * v.w);
    }
}

// ---- k6: head mean + skip, BN partial sums (deterministic two-stage) ----
// grid: 250 blocks x 256 threads; block handles 80 nodes; thread = (sub, col d)
__global__ void k_combine(const float* __restrict__ msg, const float* __restrict__ denom,
                          const float* __restrict__ P, float* __restrict__ outpre,
                          float* __restrict__ partS, float* __restrict__ partQ) {
    __shared__ float s1[256], s2[256];
    const int d   = threadIdx.x & 63;
    const int sub = threadIdx.x >> 6;
    const int nb  = blockIdx.x * 80;
    float ls = 0.f, lq = 0.f;
    for (int n = nb + sub; n < nb + 80; n += 4) {
        float acc = 0.f;
#pragma unroll
        for (int h = 0; h < HEADS; ++h) {
            float den = denom[n * HEADS + h];
            float mv  = msg[(size_t)n * HD + h * DH + d];
            acc += (den > 0.f) ? (mv / den) : 0.f;   // isolated nodes -> 0
        }
        float o = acc * 0.25f + P[(size_t)n * COLS + OFF_S + d];
        outpre[(size_t)n * DH + d] = o;
        ls += o; lq += o * o;
    }
    s1[threadIdx.x] = ls; s2[threadIdx.x] = lq;
    __syncthreads();
    if (sub == 0) {
        partS[blockIdx.x * 64 + d] = s1[d] + s1[64 + d] + s1[128 + d] + s1[192 + d];
        partQ[blockIdx.x * 64 + d] = s2[d] + s2[64 + d] + s2[128 + d] + s2[192 + d];
    }
}

// ---- k7: final BN stats ----
__global__ void k_stats(const float* __restrict__ partS, const float* __restrict__ partQ,
                        float* __restrict__ stats) {
    int d = threadIdx.x;
    if (d >= 64) return;
    float S = 0.f, Q2 = 0.f;
    for (int b = 0; b < 250; ++b) { S += partS[b * 64 + d]; Q2 += partQ[b * 64 + d]; }
    float mu  = S / (float)N_NODES;
    float var = Q2 / (float)N_NODES - mu * mu;
    stats[d]      = mu;
    stats[64 + d] = rsqrtf(var + 1e-5f);
}

// ---- k8: normalize + affine + exact GELU ----
__global__ void k_final(const float* __restrict__ outpre, const float* __restrict__ stats,
                        const float* __restrict__ gamma, const float* __restrict__ beta,
                        float* __restrict__ out) {
    int i = blockIdx.x * blockDim.x + threadIdx.x;
    if (i >= N_NODES * DH) return;
    int d = i & 63;
    float y = (outpre[i] - stats[d]) * stats[64 + d] * gamma[d] + beta[d];
    out[i] = 0.5f * y * (1.0f + erff(y * 0.70710678118654752f));
}

extern "C" void kernel_launch(void* const* d_in, const int* in_sizes, int n_in,
                              void* d_out, int out_size, void* d_ws, size_t ws_size,
                              hipStream_t stream) {
    (void)in_sizes; (void)n_in; (void)out_size; (void)ws_size;
    const float* x     = (const float*)d_in[0];
    const int*   ei    = (const int*)  d_in[1];
    const float* Wq    = (const float*)d_in[2];
    const float* bq    = (const float*)d_in[3];
    const float* Wk    = (const float*)d_in[4];
    const float* bk    = (const float*)d_in[5];
    const float* Wv    = (const float*)d_in[6];
    const float* bv    = (const float*)d_in[7];
    const float* Ws    = (const float*)d_in[8];
    const float* bs    = (const float*)d_in[9];
    const float* gamma = (const float*)d_in[10];
    const float* beta  = (const float*)d_in[11];
    float* out = (float*)d_out;

    char* base = (char*)d_ws;
    size_t off = 0;
    auto take = [&](size_t bytes) -> char* {
        char* p = base + off;
        off += (bytes + 255) & ~(size_t)255;
        return p;
    };

    __bf16*   xb      = (__bf16*)  take((size_t)N_NODES * IN_CH * sizeof(__bf16));
    __bf16*   Wall    = (__bf16*)  take((size_t)COLS * IN_CH * sizeof(__bf16));
    float*    biasAll = (float*)   take((size_t)COLS * sizeof(float));
    float*    P       = (float*)   take((size_t)N_NODES * COLS * sizeof(float)); // [Q|K|V|skip]
    float*    scores  = (float*)   take((size_t)E_EDGES * HEADS * sizeof(float));
    unsigned* smax    = (unsigned*)take((size_t)N_NODES * HEADS * sizeof(unsigned));
    float*    denom   = (float*)   take((size_t)N_NODES * HEADS * sizeof(float));
    float*    msg     = (float*)   take((size_t)N_NODES * HD * sizeof(float));
    float*    outpre  = (float*)   take((size_t)N_NODES * DH * sizeof(float));
    float*    partS   = (float*)   take((size_t)250 * 64 * sizeof(float));
    float*    partQ   = (float*)   take((size_t)250 * 64 * sizeof(float));
    float*    stats   = (float*)   take((size_t)128 * sizeof(float));

    // init accumulators (must re-run every call: harness does not re-poison)
    k_init<<<(N_NODES * HD + 255) / 256, 256, 0, stream>>>(msg, denom, smax);
    // convert inputs
    k_conv_x<<<(N_NODES * IN_CH + 255) / 256, 256, 0, stream>>>(x, xb);
    k_build_w<<<(COLS * IN_CH + 255) / 256, 256, 0, stream>>>(Wq, Wk, Wv, Ws, bq, bk, bv, bs,
                                                              Wall, biasAll);
    // fused projections (WMMA)
    k_gemm<<<N_NODES / 16, 128, 0, stream>>>(xb, Wall, biasAll, P);
    // edge phase
    k_scores<<<(E_EDGES * HEADS + 255) / 256, 256, 0, stream>>>(ei, P, scores, smax);
    k_agg<<<(E_EDGES * HEADS + 255) / 256, 256, 0, stream>>>(ei, scores, smax, P, denom, msg);
    // node phase
    k_combine<<<250, 256, 0, stream>>>(msg, denom, P, outpre, partS, partQ);
    k_stats<<<1, 64, 0, stream>>>(partS, partQ, stats);
    k_final<<<(N_NODES * DH + 255) / 256, 256, 0, stream>>>(outpre, stats, gamma, beta, out);
}
